// SAMixtureConv2d_9543417332430
// MI455X (gfx1250) — matrix operand order repaired and checked by reading, not compiled, not run on hardware
//
#include <hip/hip_runtime.h>
#include <hip/hip_bf16.h>
#include <cstdint>

// ---------------- problem constants ----------------
#define B_     2
#define CIN    64
#define COUT   64
#define H_     96
#define W_     96
#define KH_    7
#define KW_    7
#define PAD_   3
#define MIX_   4
#define REL_   32
#define PH_    (H_ + 2*PAD_)        // 102
#define PW_    (W_ + 2*PAD_)        // 102
#define PHW_   (PH_*PW_)            // 10404
#define FH_    96
#define FW_    96
#define NN     (B_*PHW_)            // 20808  (GEMM N)
#define NNP    20816                // padded to /16
#define NTILES (NNP/16)             // 1301
#define MROWS  384                  // Wq(64)+Wk(64)+Wv(4*64)
#define MTILES (MROWS/16)           // 24

typedef __attribute__((ext_vector_type(16))) __bf16 v16bf;
typedef __attribute__((ext_vector_type(8)))  float  v8f;
typedef __attribute__((ext_vector_type(4)))  unsigned int v4u;
typedef __attribute__((ext_vector_type(8)))  int  v8i;
typedef __attribute__((ext_vector_type(4)))  int  v4i;

// ---------------- prep: pad x, convert to bf16, layout [n][k] ----------------
__global__ void prep_x_kernel(const float* __restrict__ x, __bf16* __restrict__ xb) {
    int idx = blockIdx.x * blockDim.x + threadIdx.x;
    if (idx >= NNP * CIN) return;
    int n = idx >> 6;
    int c = idx & 63;
    float v = 0.0f;
    if (n < NN) {
        int b  = n / PHW_;
        int r  = n % PHW_;
        int ph = r / PW_;
        int pw = r % PW_;
        int h  = ph - PAD_;
        int w  = pw - PAD_;
        if (h >= 0 && h < H_ && w >= 0 && w < W_)
            v = x[((b * CIN + c) * H_ + h) * W_ + w];
    }
    xb[idx] = (__bf16)v;
}

// ---------------- prep: stack Wq|Wk|Wv rows and swizzle into WMMA A-layout ----
__global__ void prep_w_kernel(const float* __restrict__ Wq,
                              const float* __restrict__ Wk,
                              const float* __restrict__ Wv,
                              __bf16* __restrict__ Wsw) {
    int t = blockIdx.x * blockDim.x + threadIdx.x;
    if (t >= MTILES * 2 * 32 * 16) return;
    int e    = t & 15;
    int lane = (t >> 4) & 31;
    int kc   = (t >> 9) & 1;
    int mt   = t >> 10;
    int row  = mt * 16 + (lane & 15);
    int kb   = (lane < 16) ? 0 : 16;
    int k    = kc * 32 + kb + e;
    float v;
    if (row < 64)        v = Wq[row * CIN + k];
    else if (row < 128)  v = Wk[(row - 64) * CIN + k];
    else {
        int m = (row - 128) >> 6;
        int c = (row - 128) & 63;
        v = Wv[(m * COUT + c) * CIN + k];
    }
    Wsw[t] = (__bf16)v;
}

// ---------------- mixture gating: p_abm[m][ph][pw], softmax over m -----------
__global__ void pabm_kernel(const float* __restrict__ emb_x,
                            const float* __restrict__ emb_y,
                            const float* __restrict__ emb_m,
                            float* __restrict__ pabm) {
    int idx = blockIdx.x * blockDim.x + threadIdx.x;
    if (idx >= PHW_) return;
    int ph = idx / PW_;
    int pw = idx % PW_;
    float s[MIX_];
#pragma unroll
    for (int m = 0; m < MIX_; ++m) {
        float a = 0.0f;
        for (int c = 0; c < COUT; ++c) {
            float em = emb_m[m * COUT + c];
            a += em * (emb_x[c * PW_ + pw] + emb_y[c * PH_ + ph]);
        }
        s[m] = a;
    }
    float mx = s[0];
#pragma unroll
    for (int m = 1; m < MIX_; ++m) mx = fmaxf(mx, s[m]);
    float sum = 0.0f;
#pragma unroll
    for (int m = 0; m < MIX_; ++m) { s[m] = __expf(s[m] - mx); sum += s[m]; }
    float inv = 1.0f / sum;
#pragma unroll
    for (int m = 0; m < MIX_; ++m) pabm[m * PHW_ + idx] = s[m] * inv;
}

// ---------------- stacked projection GEMM via WMMA bf16 ----------------------
__global__ void __launch_bounds__(32)
gemm_kernel(const __bf16* __restrict__ Wsw, const __bf16* __restrict__ xb,
            float* __restrict__ P) {
    int lane = threadIdx.x;
    int nt   = blockIdx.x;
    int mg   = blockIdx.y;
    int col  = lane & 15;
    int hi   = lane >> 4;
    int n    = nt * 16 + col;

    const v16bf b0 = *(const v16bf*)(xb + n * CIN +  0 + hi * 16);
    const v16bf b1 = *(const v16bf*)(xb + n * CIN + 32 + hi * 16);

#pragma unroll
    for (int mi = 0; mi < 4; ++mi) {
        int mt = mg * 4 + mi;
        const v16bf a0 = *(const v16bf*)(Wsw + ((mt * 2 + 0) * 32 + lane) * 16);
        const v16bf a1 = *(const v16bf*)(Wsw + ((mt * 2 + 1) * 32 + lane) * 16);
        v8f acc = {};
        acc = __builtin_amdgcn_wmma_f32_16x16x32_bf16(false, a0, false, b0,
                                                      (short)0, acc, false, false);
        acc = __builtin_amdgcn_wmma_f32_16x16x32_bf16(false, a1, false, b1,
                                                      (short)0, acc, false, false);
#pragma unroll
        for (int r = 0; r < 8; ++r) {
            int row = mt * 16 + r + hi * 8;
            P[row * NNP + n] = acc[r];
        }
    }
}

// ------- mixture combine: vv[n][c] bf16, plus vk[n][c] bf16 re-layout --------
__global__ void mix_kernel(const float* __restrict__ P,
                           const float* __restrict__ pabm,
                           __bf16* __restrict__ vvb,
                           __bf16* __restrict__ vkb) {
    int idx = blockIdx.x * blockDim.x + threadIdx.x;
    if (idx >= NN * COUT) return;
    int n   = idx >> 6;
    int c   = idx & 63;
    int pos = n % PHW_;
    float acc = 0.0f;
#pragma unroll
    for (int m = 0; m < MIX_; ++m)
        acc += pabm[m * PHW_ + pos] * P[(128 + m * COUT + c) * NNP + n];
    vvb[idx] = (__bf16)acc;
    vkb[idx] = (__bf16)P[(64 + c) * NNP + n];   // vk in [n][c] bf16 for staging
}

// ---------------- windowed attention: 16x16 output tile per block ------------
#define TS    16
#define WTS   (TS + KH_ - 1)        // 22
#define WPOS  (WTS * WTS)           // 484
#define ROWE  (WTS * COUT)          // 1408 bf16 elems per window row
#define ROWB  (ROWE * 2)            // 2816 bytes per window row
#define T16   (WPOS * COUT / 8)     // 3872 16-byte chunks in the window

__global__ void __launch_bounds__(256)
attn_kernel(const float* __restrict__ P, const __bf16* __restrict__ vvb,
            const __bf16* __restrict__ vkb,
            const float* __restrict__ relx, const float* __restrict__ rely,
            const float* __restrict__ bias, float* __restrict__ out) {
    __shared__ __bf16 tile[WPOS * COUT];      // 61952 B, reused k -> v
    __shared__ float  s_relx[REL_ * KW_];
    __shared__ float  s_rely[REL_ * KH_];

    int tid = threadIdx.x;
    int b   = blockIdx.z;
    int h0  = blockIdx.y * TS;
    int w0  = blockIdx.x * TS;
    int ty  = tid >> 4;
    int tx  = tid & 15;
    int h   = h0 + ty;
    int w   = w0 + tx;

    if (tid < 2 * REL_ * KW_) {
        if (tid < REL_ * KW_) s_relx[tid] = relx[tid];
        else                  s_rely[tid - REL_ * KW_] = rely[tid - REL_ * KW_];
    }

    unsigned lds_base = (unsigned)(uintptr_t)(&tile[0]);   // low 32b = LDS offset

    // ---- phase 1: stage k window via Tensor Data Mover (wave 0 issues) ----
    // 2D tile: 22 rows x 2816 B, row stride PW_*COUT*2 B, data_size = 4 B.
    if (tid < 32) {
        const __bf16* gsrc = vkb + (size_t)(b * PHW_ + h0 * PW_ + w0) * COUT;
        unsigned long long ga = (unsigned long long)(uintptr_t)gsrc;
        const int td = 0x40000000;                   // huge tensor dims: no clip
        v4u g0;
        g0[0] = 1u;                                  // count=1, no gather
        g0[1] = lds_base;                            // lds_addr
        g0[2] = (unsigned)(ga & 0xffffffffu);        // global_addr[31:0]
        g0[3] = (unsigned)((ga >> 32) & 0x1ffffffu) | (2u << 30);  // type=2
        v8i g1;
        g1[0] = (2 << 16);                                        // data_size=4B
        g1[1] = (int)((td & 0xffff) << 16);                       // tensor_dim0 lo16
        g1[2] = (int)(((unsigned)td >> 16) | ((td & 0xffff) << 16)); // td0 hi / td1 lo
        g1[3] = (int)(((unsigned)td >> 16) | ((ROWB / 4) << 16)); // td1 hi | tile_dim0
        g1[4] = WTS;                                              // tile_dim1=22
        g1[5] = PW_ * COUT * 2 / 4;                               // dim0 stride (dwords)
        g1[6] = 0;
        g1[7] = 0;
        v4i g2 = {};
        v4i g3 = {};
        v8i g4 = {};
        __builtin_amdgcn_tensor_load_to_lds(g0, g1, g2, g3, g4, 0);
        __builtin_amdgcn_s_wait_tensorcnt(0);
    }
    __syncthreads();

    // prefetch phase-2 v window rows toward L2 while we compute logits
    if (tid < WTS)
        __builtin_prefetch(vvb + (size_t)(b * PHW_ + (h0 + tid) * PW_ + w0) * COUT, 0, 0);

    // relative-position terms from q
    int nq = b * PHW_ + (h + PAD_) * PW_ + (w + PAD_);
    float qx[KW_] = {0,0,0,0,0,0,0};
    float qy[KH_] = {0,0,0,0,0,0,0};
    for (int c = 0; c < REL_; ++c) {
        float qa = P[c * NNP + nq];
        float qb = P[(c + REL_) * NNP + nq];
#pragma unroll
        for (int j = 0; j < KW_; ++j) {
            qx[j] += qa * s_relx[c * KW_ + j];
            qy[j] += qb * s_rely[c * KH_ + j];
        }
    }
    float lg[KH_ * KW_];
#pragma unroll
    for (int o = 0; o < KH_ * KW_; ++o) lg[o] = qx[o % KW_] + qy[o / KW_];

    // q . k logits (o-loop unrolled -> lg stays in registers)
    for (int c = 0; c < COUT; ++c) {
        float qc = P[c * NNP + nq];
#pragma unroll
        for (int o = 0; o < KH_ * KW_; ++o) {
            int i = o / KW_, j = o % KW_;
            lg[o] += qc * (float)tile[((ty + i) * WTS + tx + j) * COUT + c];
        }
    }

    // softmax over 49
    float mx = lg[0];
#pragma unroll
    for (int o = 1; o < KH_ * KW_; ++o) mx = fmaxf(mx, lg[o]);
    float sum = 0.0f;
#pragma unroll
    for (int o = 0; o < KH_ * KW_; ++o) { lg[o] = __expf(lg[o] - mx); sum += lg[o]; }
    float inv = 1.0f / sum;
#pragma unroll
    for (int o = 0; o < KH_ * KW_; ++o) lg[o] *= inv;

    __syncthreads();   // all k reads complete before overwrite

    // ---- phase 2: stage v window via async global->LDS b128 copies ----
    for (int t = tid; t < T16; t += 256) {
        int row = t / (ROWB / 16);          // 0..21
        int o16 = t % (ROWB / 16);          // 16-byte chunk within row
        const __bf16* g = vvb + (size_t)(b * PHW_ + (h0 + row) * PW_ + w0) * COUT + o16 * 8;
        unsigned long long ga = (unsigned long long)(uintptr_t)g;
        unsigned laddr = lds_base + (unsigned)(row * ROWB + o16 * 16);
        asm volatile("global_load_async_to_lds_b128 %0, %1, off"
                     :: "v"(laddr), "v"(ga) : "memory");
    }
    asm volatile("s_wait_asynccnt 0x0" ::: "memory");
    __syncthreads();

    for (int c = 0; c < COUT; ++c) {
        float acc = 0.0f;
#pragma unroll
        for (int o = 0; o < KH_ * KW_; ++o) {
            int i = o / KW_, j = o % KW_;
            acc += lg[o] * (float)tile[((ty + i) * WTS + tx + j) * COUT + c];
        }
        out[((b * COUT + c) * FH_ + h) * FW_ + w] = acc + bias[c];
    }
}

// ---------------- launch ----------------
extern "C" void kernel_launch(void* const* d_in, const int* in_sizes, int n_in,
                              void* d_out, int out_size, void* d_ws, size_t ws_size,
                              hipStream_t stream) {
    const float* x    = (const float*)d_in[0];
    const float* Wq   = (const float*)d_in[1];
    const float* Wk   = (const float*)d_in[2];
    const float* Wv   = (const float*)d_in[3];
    const float* relx = (const float*)d_in[4];
    const float* rely = (const float*)d_in[5];
    const float* ex   = (const float*)d_in[6];
    const float* ey   = (const float*)d_in[7];
    const float* em   = (const float*)d_in[8];
    const float* bias = (const float*)d_in[9];
    float* out = (float*)d_out;

    char* ws = (char*)d_ws;
    size_t off = 0;
    __bf16* xb  = (__bf16*)(ws + off); off += (size_t)NNP * CIN * 2;
    __bf16* Wsw = (__bf16*)(ws + off); off += (size_t)MTILES * 2 * 32 * 16 * 2;
    off = (off + 255) & ~(size_t)255;
    float*  P    = (float*)(ws + off); off += (size_t)MROWS * NNP * 4;
    float*  pabm = (float*)(ws + off); off += (size_t)MIX_ * PHW_ * 4;
    off = (off + 255) & ~(size_t)255;
    __bf16* vvb  = (__bf16*)(ws + off); off += (size_t)NN * COUT * 2;
    off = (off + 255) & ~(size_t)255;
    __bf16* vkb  = (__bf16*)(ws + off); off += (size_t)NN * COUT * 2;

    prep_x_kernel<<<(NNP * CIN + 255) / 256, 256, 0, stream>>>(x, xb);
    prep_w_kernel<<<(MTILES * 2 * 32 * 16 + 255) / 256, 256, 0, stream>>>(Wq, Wk, Wv, Wsw);
    pabm_kernel<<<(PHW_ + 255) / 256, 256, 0, stream>>>(ex, ey, em, pabm);
    gemm_kernel<<<dim3(NTILES, MROWS / 64), 32, 0, stream>>>(Wsw, xb, P);
    mix_kernel<<<(NN * COUT + 255) / 256, 256, 0, stream>>>(P, pabm, vvb, vkb);
    attn_kernel<<<dim3(FW_ / TS, FH_ / TS, B_), 256, 0, stream>>>(P, vvb, vkb, relx, rely, bias, out);
}
